// GCNGAT_38448547233923
// MI455X (gfx1250) — matrix operand order repaired and checked by reading, not compiled, not run on hardware
//
#include <hip/hip_runtime.h>
#include <math.h>

// ---------------- problem constants (match reference) ----------------
#define NN      50000
#define EE      800000
#define ETOT    (EE + NN)          // edges + self loops
#define INC     128
#define HCC     64
#define NHEADS  4
#define OUTC    32
#define NEG_SLOPE 0.2f

typedef __attribute__((ext_vector_type(16))) __bf16 v16bf;
typedef __attribute__((ext_vector_type(8)))  float  v8f;
typedef __attribute__((ext_vector_type(4)))  int    v4i;

// ---- gfx1250 async global->LDS path (guarded; sync fallback otherwise) ----
#if defined(__has_builtin)
#  if __has_builtin(__builtin_amdgcn_global_load_async_to_lds_b128)
#    define HAS_ASYNC_LDS 1
#  endif
#endif
#ifndef HAS_ASYNC_LDS
#  define HAS_ASYNC_LDS 0
#endif

static __device__ __forceinline__ void wait_async0() {
#if defined(__has_builtin) && __has_builtin(__builtin_amdgcn_s_wait_asynccnt)
  __builtin_amdgcn_s_wait_asynccnt(0);
#else
  asm volatile("s_wait_asynccnt 0" ::: "memory");
#endif
}

static __device__ __forceinline__ float elu_f(float x) {
  return x > 0.f ? x : (expf(x) - 1.f);
}
static __device__ __forceinline__ float lrelu_f(float x) {
  return x > 0.f ? x : NEG_SLOPE * x;
}
// round-to-nearest-even f32 -> bf16 (bit manipulation)
static __device__ __forceinline__ unsigned short f2bf(float f) {
  unsigned int u = __float_as_uint(f);
  u += 0x7FFFu + ((u >> 16) & 1u);
  return (unsigned short)(u >> 16);
}

// ---------------- bf16 converters (one-shot prep passes) ----------------
__global__ void k_cvt_bf16(const float* __restrict__ s, unsigned short* __restrict__ d,
                           long long n) {
  long long i = (long long)blockIdx.x * blockDim.x + threadIdx.x;
  if (i < n) d[i] = f2bf(s[i]);
}
// W[K x Nd] fp32 row-major -> Wt[Nd x K] bf16 (transposed, K-contiguous)
__global__ void k_cvt_wt_bf16(const float* __restrict__ W, unsigned short* __restrict__ Wt,
                              int K, int Nd) {
  long long i = (long long)blockIdx.x * blockDim.x + threadIdx.x;
  if (i >= (long long)K * Nd) return;
  int n = (int)(i % Nd);
  int k = (int)(i / Nd);
  Wt[(size_t)n * K + k] = f2bf(W[i]);
}

// ---------------- WMMA GEMM ----------------
// Out[M x Nd] = Abf[M x K](bf16) @ Wt[Nd x K](bf16, pre-transposed)
// block = 256 (8 wave32 waves). Each wave owns 16 rows x (16*NTILE) cols.
// grid = (ceil(M/128), Nd/(16*NTILE)). K multiple of 32 (K in {128,64,256}).
// B panel staged once (full K) in LDS via async b128 copies; A fragments are
// two direct global b128 loads per lane; no barriers inside the K loop.
template <int NTILE>
__global__ __launch_bounds__(256)
void k_gemm_wmma(const unsigned short* __restrict__ Abf,
                 const unsigned short* __restrict__ Wt,
                 float* __restrict__ Out, int M, int K, int Nd) {
  __shared__ unsigned short Bs[64 * 264];    // up to 64 cols x (256 + 8 pad)
  const int KS = K + 8;                      // row stride (mult of 8 -> 16B align)
  const int tid = threadIdx.x;
  const int wv = tid >> 5, l = tid & 31, half = l >> 4, mr = l & 15;
  const int rowBlk = blockIdx.x * 128;
  const int colBlk = blockIdx.y * (16 * NTILE);

  // ---- stage B panel [16*NTILE x K] ----
  const int segs = K >> 3;                   // b128 chunks per row
  for (int c = tid; c < 16 * NTILE * segs; c += 256) {
    int n = c / segs, sg = c - n * segs;
    const unsigned short* src = Wt + (size_t)(colBlk + n) * K + sg * 8;
    unsigned short* dst = &Bs[n * KS + sg * 8];
#if HAS_ASYNC_LDS
    __builtin_amdgcn_global_load_async_to_lds_b128((v4i*)(void*)src, (v4i*)(void*)dst, 0, 0);
#else
    *(uint4*)dst = *(const uint4*)src;
#endif
  }
#if HAS_ASYNC_LDS
  wait_async0();
#endif
  __syncthreads();

  // ---- main K loop: 1 A-fragment feeds NTILE WMMAs ----
  const unsigned short* arow = Abf + (size_t)(rowBlk + wv * 16 + mr) * K;
  v8f acc[NTILE] = {};
  for (int k0 = 0; k0 < K; k0 += 32) {
    union { v16bf v; unsigned int u[8]; } fa;
    // CDNA5 16-bit A 16x32 layout: lane half 0 -> K 0..7 & 16..23, half 1 -> 8..15 & 24..31
    *(uint4*)&fa.u[0] = *(const uint4*)(arow + k0 + half * 8);
    *(uint4*)&fa.u[4] = *(const uint4*)(arow + k0 + 16 + half * 8);
#pragma unroll
    for (int nt = 0; nt < NTILE; ++nt) {
      union { v16bf v; unsigned int u[8]; } fb;
      // B 32x16 layout: lanes 0-15 col N=l hold K 0..7 & 16..23; lanes 16-31 hold K 8..15 & 24..31
      const unsigned short* bp = &Bs[(nt * 16 + mr) * KS + k0 + half * 8];
      *(uint4*)&fb.u[0] = *(const uint4*)bp;
      *(uint4*)&fb.u[4] = *(const uint4*)(bp + 16);
      acc[nt] = __builtin_amdgcn_wmma_f32_16x16x32_bf16(false, fa.v, false, fb.v,
                                                        (short)0, acc[nt], false, false);
    }
  }

  // ---- store: C layout lane l -> col mr, rows half*8 + v ----
  const int rowBase = rowBlk + wv * 16 + half * 8;
#pragma unroll
  for (int nt = 0; nt < NTILE; ++nt) {
    int col = colBlk + nt * 16 + mr;
#pragma unroll
    for (int v = 0; v < 8; ++v) {
      int r = rowBase + v;
      if (r < M) Out[(size_t)r * Nd + col] = acc[nt][v];
    }
  }
}

// ---------------- elementwise / graph kernels ----------------
__global__ void k_fill(float* __restrict__ p, float val, long long n) {
  long long i = (long long)blockIdx.x * blockDim.x + threadIdx.x;
  if (i < n) p[i] = val;
}

__global__ void k_degree(const long long* __restrict__ ei, float* __restrict__ deg) {
  long long i = (long long)blockIdx.x * blockDim.x + threadIdx.x;
  if (i >= (long long)ETOT) return;
  int d = (i < EE) ? (int)ei[EE + i] : (int)(i - EE);
  atomicAdd(&deg[d], 1.0f);
}

__global__ void k_rsqrt_inplace(float* __restrict__ p, int n) {
  int i = blockIdx.x * blockDim.x + threadIdx.x;
  if (i < n) { float d = p[i]; p[i] = d > 0.f ? rsqrtf(d) : 0.f; }
}

// GCN: agg[dst, c] += h[src, c] * dinv[src]*dinv[dst]; thread per (edge, c)
__global__ void k_gcn_aggregate(const long long* __restrict__ ei,
                                const float* __restrict__ h,
                                const float* __restrict__ dinv,
                                float* __restrict__ agg, int C) {
  long long gid = (long long)blockIdx.x * blockDim.x + threadIdx.x;
  long long tot = (long long)ETOT * C;
  if (gid >= tot) return;
  long long e = gid / C;
  int c = (int)(gid - e * C);
  int s, d;
  if (e < EE) { s = (int)ei[e]; d = (int)ei[EE + e]; } else { s = d = (int)(e - EE); }
  float norm = dinv[s] * dinv[d];
  atomicAdd(&agg[(size_t)d * C + c], h[(size_t)s * C + c] * norm);
}

// x = elu(x + b); also emit packed bf16 copy for the next layer's WMMA GEMM
__global__ void k_bias_elu(float* __restrict__ x, const float* __restrict__ b,
                           unsigned short* __restrict__ xbf, int C, long long n) {
  long long i = (long long)blockIdx.x * blockDim.x + threadIdx.x;
  if (i >= n) return;
  float v = elu_f(x[i] + b[i % C]);
  x[i] = v;
  xbf[i] = f2bf(v);
}

// per-node attention logits (wave per node)
__global__ void k_alpha(const float* __restrict__ h, const float* __restrict__ a_s,
                        const float* __restrict__ a_d, float* __restrict__ as_out,
                        float* __restrict__ ad_out, int H, int C) {
  int gid  = blockIdx.x * blockDim.x + threadIdx.x;
  int node = gid >> 5;
  int l    = gid & 31;
  if (node >= NN) return;
  for (int hh = 0; hh < H; ++hh) {
    float ps = 0.f, pd = 0.f;
    for (int c = l; c < C; c += 32) {
      float v = h[(size_t)node * H * C + hh * C + c];
      ps += v * a_s[hh * C + c];
      pd += v * a_d[hh * C + c];
    }
    for (int off = 16; off > 0; off >>= 1) {   // wave32 butterfly
      ps += __shfl_xor(ps, off, 32);
      pd += __shfl_xor(pd, off, 32);
    }
    if (l == 0) { as_out[node * H + hh] = ps; ad_out[node * H + hh] = pd; }
  }
}

static __device__ __forceinline__ void edge_sd(const long long* __restrict__ ei,
                                               long long e, int& s, int& d) {
  if (e < EE) { s = (int)ei[e]; d = (int)ei[EE + e]; } else { s = d = (int)(e - EE); }
}

__global__ void k_edge_max(const long long* __restrict__ ei,
                           const float* __restrict__ asn, const float* __restrict__ adn,
                           float* __restrict__ mx, int H) {
  long long gid = (long long)blockIdx.x * blockDim.x + threadIdx.x;
  if (gid >= (long long)ETOT * H) return;
  long long e = gid / H;
  int hh = (int)(gid - e * H);
  int s, d; edge_sd(ei, e, s, d);
  float ev = lrelu_f(asn[s * H + hh] + adn[d * H + hh]);
  atomicMax(&mx[d * H + hh], ev);
}

__global__ void k_edge_expsum(const long long* __restrict__ ei,
                              const float* __restrict__ asn, const float* __restrict__ adn,
                              const float* __restrict__ mx, float* __restrict__ sm, int H) {
  long long gid = (long long)blockIdx.x * blockDim.x + threadIdx.x;
  if (gid >= (long long)ETOT * H) return;
  long long e = gid / H;
  int hh = (int)(gid - e * H);
  int s, d; edge_sd(ei, e, s, d);
  float ev = lrelu_f(asn[s * H + hh] + adn[d * H + hh]);
  atomicAdd(&sm[d * H + hh], expf(ev - mx[d * H + hh]));
}

__global__ void k_edge_aggregate(const long long* __restrict__ ei,
                                 const float* __restrict__ h,
                                 const float* __restrict__ asn, const float* __restrict__ adn,
                                 const float* __restrict__ mx, const float* __restrict__ sm,
                                 float* __restrict__ agg, int H, int C) {
  long long gid = (long long)blockIdx.x * blockDim.x + threadIdx.x;
  int HC = H * C;
  long long tot = (long long)ETOT * HC;
  if (gid >= tot) return;
  long long e = gid / HC;
  int c_all = (int)(gid - e * HC);
  int hh = c_all / C;
  int s, d; edge_sd(ei, e, s, d);
  float ev = lrelu_f(asn[s * H + hh] + adn[d * H + hh]);
  float ex = expf(ev - mx[d * H + hh]);
  float alpha = ex / (sm[d * H + hh] + 1e-16f);
  atomicAdd(&agg[(size_t)d * HC + c_all], h[(size_t)s * HC + c_all] * alpha);
}

// head: out = log_softmax(agg + bias) per node over OUTC=32 (wave per node)
__global__ void k_head_logsoftmax(const float* __restrict__ agg,
                                  const float* __restrict__ bias,
                                  float* __restrict__ out) {
  int gid  = blockIdx.x * blockDim.x + threadIdx.x;
  int node = gid >> 5;
  int l    = gid & 31;
  if (node >= NN) return;
  float x = agg[(size_t)node * OUTC + l] + bias[l];
  float m = x;
  for (int off = 16; off > 0; off >>= 1) m = fmaxf(m, __shfl_xor(m, off, 32));
  float ex = expf(x - m);
  float s = ex;
  for (int off = 16; off > 0; off >>= 1) s += __shfl_xor(s, off, 32);
  out[(size_t)node * OUTC + l] = x - m - logf(s);
}

// ---------------- driver ----------------
static inline int cdivll(long long a, long long b) { return (int)((a + b - 1) / b); }

extern "C" void kernel_launch(void* const* d_in, const int* in_sizes, int n_in,
                              void* d_out, int out_size, void* d_ws, size_t ws_size,
                              hipStream_t stream) {
  const float*     x   = (const float*)d_in[0];
  const long long* ei  = (const long long*)d_in[1];   // int64 [2, E]
  const float* gcn1_W = (const float*)d_in[2];
  const float* gcn1_b = (const float*)d_in[3];
  const float* gcn2_W = (const float*)d_in[4];
  const float* gcn2_b = (const float*)d_in[5];
  const float* gat1_W = (const float*)d_in[6];
  const float* gat1_as = (const float*)d_in[7];
  const float* gat1_ad = (const float*)d_in[8];
  const float* gat1_b  = (const float*)d_in[9];
  const float* gat2_W = (const float*)d_in[10];
  const float* gat2_as = (const float*)d_in[11];
  const float* gat2_ad = (const float*)d_in[12];
  const float* gat2_b  = (const float*)d_in[13];
  const float* gat3_W = (const float*)d_in[14];
  const float* gat3_as = (const float*)d_in[15];
  const float* gat3_ad = (const float*)d_in[16];
  const float* gat3_b  = (const float*)d_in[17];
  float* out = (float*)d_out;

  // workspace layout
  float* ws   = (float*)d_ws;
  float* dinv = ws;                               // N
  float* hAcc = dinv + NN;                        // N*256 (aggregation target)
  float* hT   = hAcc + (size_t)NN * 256;          // N*256 (GEMM output / features)
  float* asn  = hT + (size_t)NN * 256;            // N*4
  float* adn  = asn + (size_t)NN * NHEADS;
  float* mx   = adn + (size_t)NN * NHEADS;
  float* sm   = mx + (size_t)NN * NHEADS;
  unsigned short* hbf = (unsigned short*)(sm + (size_t)NN * NHEADS); // N*256 bf16
  unsigned short* wtbf = hbf + (size_t)NN * 256;  // 102400 bf16 (all weights, transposed)
  unsigned short* w1t = wtbf;                     // 64 x 128
  unsigned short* w2t = w1t + 128 * 64;           // 64 x 64
  unsigned short* w3t = w2t + 64 * 64;            // 256 x 64
  unsigned short* w4t = w3t + 64 * 256;           // 256 x 256
  unsigned short* w5t = w4t + 256 * 256;          // 32 x 256

  const int B = 256;
  auto fill = [&](float* p, float v, long long n) {
    k_fill<<<cdivll(n, B), B, 0, stream>>>(p, v, n);
  };
  auto cvt_wt = [&](const float* W, unsigned short* Wt, int K, int Nd) {
    k_cvt_wt_bf16<<<cdivll((long long)K * Nd, B), B, 0, stream>>>(W, Wt, K, Nd);
  };
  auto gemm = [&](const unsigned short* Abf, const unsigned short* Wt, float* O,
                  int M, int K, int Nd) {
    if (Nd % 64 == 0)
      k_gemm_wmma<4><<<dim3(cdivll(M, 128), Nd / 64), 256, 0, stream>>>(Abf, Wt, O, M, K, Nd);
    else
      k_gemm_wmma<2><<<dim3(cdivll(M, 128), Nd / 32), 256, 0, stream>>>(Abf, Wt, O, M, K, Nd);
  };

  // one-shot preparation: bf16 transposed weights, bf16 input features
  cvt_wt(gcn1_W, w1t, INC, HCC);
  cvt_wt(gcn2_W, w2t, HCC, HCC);
  cvt_wt(gat1_W, w3t, HCC, NHEADS * HCC);
  cvt_wt(gat2_W, w4t, NHEADS * HCC, NHEADS * HCC);
  cvt_wt(gat3_W, w5t, NHEADS * HCC, OUTC);
  k_cvt_bf16<<<cdivll((long long)NN * INC, B), B, 0, stream>>>(x, hbf, (long long)NN * INC);

  // degree -> dinv (shared by both GCN layers)
  fill(dinv, 0.f, NN);
  k_degree<<<cdivll((long long)ETOT, B), B, 0, stream>>>(ei, dinv);
  k_rsqrt_inplace<<<cdivll(NN, B), B, 0, stream>>>(dinv, NN);

  // ---- GCN1 ----
  gemm(hbf, w1t, hT, NN, INC, HCC);
  fill(hAcc, 0.f, (long long)NN * HCC);
  k_gcn_aggregate<<<cdivll((long long)ETOT * HCC, B), B, 0, stream>>>(ei, hT, dinv, hAcc, HCC);
  k_bias_elu<<<cdivll((long long)NN * HCC, B), B, 0, stream>>>(hAcc, gcn1_b, hbf, HCC, (long long)NN * HCC);

  // ---- GCN2 ----
  gemm(hbf, w2t, hT, NN, HCC, HCC);
  fill(hAcc, 0.f, (long long)NN * HCC);
  k_gcn_aggregate<<<cdivll((long long)ETOT * HCC, B), B, 0, stream>>>(ei, hT, dinv, hAcc, HCC);
  k_bias_elu<<<cdivll((long long)NN * HCC, B), B, 0, stream>>>(hAcc, gcn2_b, hbf, HCC, (long long)NN * HCC);

  // ---- generic GAT layer ----
  auto gat = [&](const unsigned short* wt, int Kin, const float* a_s, const float* a_d,
                 const float* bias, int H, int C, bool elu_act) {
    const int HC = H * C;
    gemm(hbf, wt, hT, NN, Kin, HC);
    k_alpha<<<cdivll((long long)NN * 32, B), B, 0, stream>>>(hT, a_s, a_d, asn, adn, H, C);
    fill(mx, -3.0e38f, (long long)NN * H);
    fill(sm, 0.f, (long long)NN * H);
    k_edge_max<<<cdivll((long long)ETOT * H, B), B, 0, stream>>>(ei, asn, adn, mx, H);
    k_edge_expsum<<<cdivll((long long)ETOT * H, B), B, 0, stream>>>(ei, asn, adn, mx, sm, H);
    fill(hAcc, 0.f, (long long)NN * HC);
    k_edge_aggregate<<<cdivll((long long)ETOT * HC, B), B, 0, stream>>>(ei, hT, asn, adn, mx, sm, hAcc, H, C);
    if (elu_act)
      k_bias_elu<<<cdivll((long long)NN * HC, B), B, 0, stream>>>(hAcc, bias, hbf, HC, (long long)NN * HC);
  };

  gat(w3t, HCC,          gat1_as, gat1_ad, gat1_b, NHEADS, HCC, true);   // GAT1
  gat(w4t, NHEADS * HCC, gat2_as, gat2_ad, gat2_b, NHEADS, HCC, true);   // GAT2
  gat(w5t, NHEADS * HCC, gat3_as, gat3_ad, gat3_b, 1, OUTC, false);      // GAT3 head

  k_head_logsoftmax<<<cdivll((long long)NN * 32, B), B, 0, stream>>>(hAcc, gat3_b, out);
}
